// CorrOptDiMPQReg_85023172592330
// MI455X (gfx1250) — matrix operand order repaired and checked by reading, not compiled, not run on hardware
//
#include <hip/hip_runtime.h>

// ---------------------------------------------------------------------------
// CDNA5 (gfx1250) implementation of the DiMP filter optimizer.
// Heavy grouped convs are per-sequence GEMMs executed with
// v_wmma_f32_16x16x32_bf16 (wave32), im2col gathered on the fly into LDS.
// A-tiles use GLOBAL_LOAD_ASYNC_TO_LDS (ASYNCcnt) when the toolchain has it.
// ---------------------------------------------------------------------------

typedef __attribute__((ext_vector_type(16))) __bf16 v16bf;
typedef __attribute__((ext_vector_type(8)))  float  v8f;
typedef __attribute__((ext_vector_type(4)))  int    v4i;

#define S_   16
#define C_   256
#define H_   20
#define W_   20
#define P_   400          // H*W
#define F_   400          // num filters = H*W
#define K9_  2304         // C*9
#define NB_  10
#define PP_  (F_ * P_)    // 160000 map elements

#if defined(__has_builtin)
#  if __has_builtin(__builtin_amdgcn_global_load_async_to_lds_b128) && \
      __has_builtin(__builtin_amdgcn_s_wait_asynccnt)
#    define USE_ASYNC_LDS 1
#  endif
#endif
#ifndef USE_ASYNC_LDS
#  define USE_ASYNC_LDS 0
#endif

#if USE_ASYNC_LDS
typedef __attribute__((address_space(1))) v4i GV4;   // global 128-bit chunk
typedef __attribute__((address_space(3))) v4i LV4;   // LDS 128-bit chunk
#endif

__device__ __forceinline__ unsigned short f2bf(float f) {
    unsigned int u = __float_as_uint(f);
    unsigned int r = (u + 0x7FFFu + ((u >> 16) & 1u)) >> 16;   // RNE
    return (unsigned short)r;
}

union Frag16 { uint4 q[2]; v16bf v; };

// ---------------------------------------------------------------------------
// WMMA GEMM:  D[s][m][n] = sum_k A[s][m][k] * B_gather[s][k][n]
//   MODE 0 (apply_filter):   k=(c,ky,kx), n=(y,x)  -> B = im2col(Feat)
//   MODE 1 (feat_transpose): k=(y,x), n=(c,ky,kx)  -> B = im2col(Feat)^T
//   ep 0: D = acc   ep 1: D = acc + reg_w*Filt   ep 2: D += acc
// Requires K % 8 == 0 (true: 2304 and 400).
// ---------------------------------------------------------------------------
template <int MODE>
__global__ __launch_bounds__(256) void k_gemm_bf16(
    const unsigned short* __restrict__ A,     // S x M x K (bf16 bits)
    const unsigned short* __restrict__ Feat,  // S x C x H x W (bf16 bits)
    float* __restrict__ D,                    // S x M x N
    const float* __restrict__ FiltF32,        // for ep==1
    const float* __restrict__ filter_reg,     // [1], for ep==1
    int M, int N, int K, int ep)
{
    __shared__ unsigned short As[64][40];   // m-major, +8 pad (16B-aligned rows)
    __shared__ unsigned short Bs[64][40];   // n-major, +8 pad

    const int s  = blockIdx.z;
    const int bm = blockIdx.y * 64;
    const int bn = blockIdx.x * 64;
    const int t  = threadIdx.x;

    // A staging: thread t covers row t/4, 8 contiguous k at (t%4)*8
    const int ar = t >> 2;
    const int ac = (t & 3) << 3;
    const bool mok = (bm + ar) < M;
    const unsigned short* Arow =
        A + ((size_t)s * M + (mok ? (bm + ar) : 0)) * K;   // clamped-safe base

    // B staging: thread t gathers 8 k-values for column n_local = t%64
    const int bnl = t & 63;
    const int bkl = (t >> 6) << 3;
    const int nn  = bn + bnl;

    const int lane = t & 31;
    const int wave = t >> 5;
    const int half = lane >> 4;
    const int l16  = lane & 15;
    const int mi   = wave & 3;          // m-fragment 0..3
    const int nj   = (wave >> 2) << 1;  // n-fragments nj, nj+1

    v8f acc0 = {0.f,0.f,0.f,0.f,0.f,0.f,0.f,0.f};
    v8f acc1 = acc0;

    for (int k0 = 0; k0 < K; k0 += 32) {
        // ---- stage A tile (64x32 bf16), branch-free / async ----
        const bool aok = mok && (k0 + ac + 8 <= K);
#if USE_ASYNC_LDS
        if (aok) {
            __builtin_amdgcn_global_load_async_to_lds_b128(
                (GV4*)&Arow[k0 + ac], (LV4*)(void*)&As[ar][ac], 0, 0);
        } else {
            uint4 z = {0u, 0u, 0u, 0u};
            *(uint4*)&As[ar][ac] = z;
        }
#else
        {
            uint4 d = *(const uint4*)&Arow[aok ? (k0 + ac) : 0];
            if (!aok) { d.x = 0u; d.y = 0u; d.z = 0u; d.w = 0u; }
            *(uint4*)&As[ar][ac] = d;
        }
#endif
        // ---- stage B tile (32x64 bf16) via branch-free im2col gather ----
#pragma unroll
        for (int e = 0; e < 8; ++e) {
            const int kk   = k0 + bkl + e;
            const int pRaw = (MODE == 0) ? nn : kk;   // decodes (y,x)
            const int cRaw = (MODE == 0) ? kk : nn;   // decodes (c,ky,kx)
            const int pc = min(pRaw, P_ - 1);
            const int cc = min(cRaw, K9_ - 1);
            const int c  = cc / 9;
            const int r9 = cc - c * 9;
            const int ky = r9 / 3;
            const int kx = r9 - ky * 3;
            const int py = pc / W_;
            const int y  = py + ky - 1;
            const int x  = (pc - py * W_) + kx - 1;
            const bool inb = (kk < K) & (nn < N) &
                             (y >= 0) & (y < H_) & (x >= 0) & (x < W_);
            const int yc = min(max(y, 0), H_ - 1);
            const int xc = min(max(x, 0), W_ - 1);
            unsigned short v = Feat[((s * C_ + c) * H_ + yc) * W_ + xc];
            Bs[bnl][bkl + e] = inb ? v : (unsigned short)0;
        }
#if USE_ASYNC_LDS
        __builtin_amdgcn_s_wait_asynccnt(0);
#endif
        __syncthreads();

        // ---- fragments per documented CDNA5 wave32 layouts ----
        Frag16 af, b0, b1;
        af.q[0] = *(const uint4*)&As[mi * 16 + l16][half * 8];
        af.q[1] = *(const uint4*)&As[mi * 16 + l16][16 + half * 8];
        b0.q[0] = *(const uint4*)&Bs[nj * 16 + l16][half * 8];
        b0.q[1] = *(const uint4*)&Bs[nj * 16 + l16][16 + half * 8];
        b1.q[0] = *(const uint4*)&Bs[(nj + 1) * 16 + l16][half * 8];
        b1.q[1] = *(const uint4*)&Bs[(nj + 1) * 16 + l16][16 + half * 8];

        acc0 = __builtin_amdgcn_wmma_f32_16x16x32_bf16(
                   false, af.v, false, b0.v, (short)0, acc0, false, false);
        acc1 = __builtin_amdgcn_wmma_f32_16x16x32_bf16(
                   false, af.v, false, b1.v, (short)0, acc1, false, false);
        __syncthreads();
    }

    float rw = 0.f;
    if (ep == 1) {
        float fr = filter_reg[0];
        rw = fmaxf(fr * fr, 1e-10f);
    }
#pragma unroll
    for (int r = 0; r < 8; ++r) {
        int m = bm + mi * 16 + r + half * 8;   // C/D layout: M = vgpr + 8*half
        if (m >= M) continue;
#pragma unroll
        for (int j = 0; j < 2; ++j) {
            int n = bn + (nj + j) * 16 + l16;  // N = lane % 16 within fragment
            if (n >= N) continue;
            size_t idx = ((size_t)s * M + m) * N + n;
            float val = (j == 0) ? acc0[r] : acc1[r];
            if (ep == 1)      D[idx] = val + rw * FiltF32[idx];
            else if (ep == 2) D[idx] += val;
            else              D[idx] = val;
        }
    }
}

// ---------------------------------------------------------------------------
// Distance-map + unfold fusion: label / sigmoid(mask) / spatial maps (F x P)
// ---------------------------------------------------------------------------
__global__ void k_maps(const float* __restrict__ wl, const float* __restrict__ wm,
                       const float* __restrict__ wsp,
                       float* __restrict__ lab, float* __restrict__ msk,
                       float* __restrict__ sp)
{
    int i = blockIdx.x * blockDim.x + threadIdx.x;
    if (i >= PP_) return;
    int f = i / P_, p = i - f * P_;
    int fi = f / W_, fj = f - fi * W_;
    int y  = p / W_, x  = p - y * W_;
    float dy = (float)(y - fi), dx = (float)(x - fj);
    float bd = sqrtf(dy * dy + dx * dx) * 2.0f;    // dist / 0.5
    float L = 0.f, Mm = 0.f, Sp = 0.f;
    for (int b = 0; b < NB_; ++b) {
        float diff = bd - (float)b;
        float v = (b < NB_ - 1) ? fmaxf(1.f - fabsf(diff), 0.f)
                                : fminf(fmaxf(1.f + diff, 0.f), 1.f);
        L += wl[b] * v; Mm += wm[b] * v; Sp += wsp[b] * v;
    }
    lab[i] = L;
    msk[i] = 1.f / (1.f + expf(-Mm));
    sp[i]  = Sp;
}

__global__ void k_f32_to_bf16(const float* __restrict__ src,
                              unsigned short* __restrict__ dst, int n)
{
    int i = blockIdx.x * blockDim.x + threadIdx.x;
    if (i < n) dst[i] = f2bf(src[i]);
}

// LeakyReluPar residual:  mapped = mask_deriv * spatial^2 * (act - label)
__global__ void k_postscores(const float* __restrict__ scores,
                             const float* __restrict__ lab,
                             const float* __restrict__ msk,
                             const float* __restrict__ sp,
                             unsigned short* __restrict__ mappedb)
{
    int i = blockIdx.x * blockDim.x + threadIdx.x;
    if (i >= S_ * PP_) return;
    int fp = i % PP_;
    float m = msk[fp], l = lab[fp], w = sp[fp];
    float sc = scores[i];
    float act = (1.f - m) * 0.5f * fabsf(sc) + (1.f + m) * 0.5f * sc;
    float sgn = (sc > 0.f) ? 1.f : ((sc < 0.f) ? -1.f : 0.f);
    float md  = (1.f - m) * 0.5f * sgn + (1.f + m) * 0.5f;
    mappedb[i] = f2bf(md * w * w * (act - l));
}

// sg = spatial * mask_deriv(scores) * sg_raw   (in place)
__global__ void k_sg(float* __restrict__ sgraw, const float* __restrict__ scores,
                     const float* __restrict__ msk, const float* __restrict__ sp)
{
    int i = blockIdx.x * blockDim.x + threadIdx.x;
    if (i >= S_ * PP_) return;
    int fp = i % PP_;
    float m = msk[fp], w = sp[fp];
    float sc = scores[i];
    float sgn = (sc > 0.f) ? 1.f : ((sc < 0.f) ? -1.f : 0.f);
    float md  = (1.f - m) * 0.5f * sgn + (1.f + m) * 0.5f;
    sgraw[i] = w * md * sgraw[i];
}

// Generic 3x3 correlation over the innermost (H,W) index of (B=16, A=400, P=400),
// optional kernel flip (adjoint) and optional transposed (b,p,a) output.
__global__ void k_conv_inner(const float* __restrict__ in,
                             float* __restrict__ outF,
                             unsigned short* __restrict__ outB,
                             const float* __restrict__ w9,
                             int flip, int transpose_out)
{
    int i = blockIdx.x * blockDim.x + threadIdx.x;
    if (i >= S_ * P_ * P_) return;
    int b   = i / (P_ * P_);
    int rem = i - b * (P_ * P_);
    int a   = rem / P_;
    int p   = rem - a * P_;
    int y = p / W_, x = p - y * W_;
    const float* base = in + ((size_t)b * P_ + a) * P_;
    float sum = 0.f;
    for (int u = 0; u < 3; ++u) {
        int yy = y + u - 1;
        if (yy < 0 || yy >= H_) continue;
        for (int v = 0; v < 3; ++v) {
            int xx = x + v - 1;
            if (xx < 0 || xx >= W_) continue;
            float wv = flip ? w9[(2 - u) * 3 + (2 - v)] : w9[u * 3 + v];
            sum += wv * base[yy * W_ + xx];
        }
    }
    size_t o = transpose_out ? (((size_t)b * P_ + p) * P_ + a) : (size_t)i;
    if (outF) outF[o] = sum;
    if (outB) outB[o] = f2bf(sum);
}

// per-sequence sum of squares -> acc[s]
__global__ void k_reduce_sq(const float* __restrict__ src,
                            float* __restrict__ acc, int lenPerS)
{
    __shared__ float red[256];
    int s = blockIdx.y;
    int i = blockIdx.x * 256 + threadIdx.x;
    float v = 0.f;
    if (i < lenPerS) { float x = src[(size_t)s * lenPerS + i]; v = x * x; }
    red[threadIdx.x] = v;
    __syncthreads();
    for (int st = 128; st > 0; st >>= 1) {
        if (threadIdx.x < st) red[threadIdx.x] += red[threadIdx.x + st];
        __syncthreads();
    }
    if (threadIdx.x == 0) atomicAdd(&acc[s], red[0]);
}

__global__ void k_zero(float* __restrict__ p, int n)
{
    int i = blockIdx.x * blockDim.x + threadIdx.x;
    if (i < n) p[i] = 0.f;
}

// filt -= exp(lsl) * alpha[s] * f_grad
__global__ void k_update(float* __restrict__ filt, const float* __restrict__ fgrad,
                         const float* __restrict__ accN, const float* __restrict__ accS,
                         const float* __restrict__ accL,
                         const float* __restrict__ lsl, const float* __restrict__ fr)
{
    int i = blockIdx.x * blockDim.x + threadIdx.x;
    if (i >= S_ * F_ * K9_) return;
    int s = i / (F_ * K9_);
    float num = accN[s];
    float rw  = fmaxf(fr[0] * fr[0], 1e-10f);
    float den = fmaxf(accS[s] + accL[s] + rw * num, 1e-8f);
    float alpha = num / den;
    filt[i] -= expf(lsl[0]) * alpha * fgrad[i];
}

// ---------------------------------------------------------------------------
extern "C" void kernel_launch(void* const* d_in, const int* in_sizes, int n_in,
                              void* d_out, int out_size, void* d_ws, size_t ws_size,
                              hipStream_t stream)
{
    (void)in_sizes; (void)n_in; (void)out_size; (void)ws_size;

    const float* filt0 = (const float*)d_in[0];
    const float* feat  = (const float*)d_in[1];
    const float* tfeat = (const float*)d_in[2];
    const float* lsl   = (const float*)d_in[3];
    const float* freg  = (const float*)d_in[4];
    const float* wl    = (const float*)d_in[5];
    const float* wm    = (const float*)d_in[6];
    const float* wsp   = (const float*)d_in[7];
    const float* rw1   = (const float*)d_in[8];
    const float* rw2   = (const float*)d_in[9];
    float* filt = (float*)d_out;                       // f32 master filter

    const int SFK  = S_ * F_ * K9_;    // 14,745,600
    const int SFP  = S_ * PP_;         //  2,560,000
    const int SCHW = S_ * C_ * H_ * W_;

    char* wbase = (char*)d_ws;
    size_t off = 0;
    auto carve = [&](size_t bytes) -> void* {
        void* p = (void*)(wbase + off);
        off = (off + bytes + 255) & ~((size_t)255);
        return p;
    };
    float*          lab     = (float*)         carve((size_t)PP_ * 4);
    float*          msk     = (float*)         carve((size_t)PP_ * 4);
    float*          sp      = (float*)         carve((size_t)PP_ * 4);
    unsigned short* featb   = (unsigned short*)carve((size_t)SCHW * 2);
    unsigned short* tfeatb  = (unsigned short*)carve((size_t)SCHW * 2);
    unsigned short* filtb   = (unsigned short*)carve((size_t)SFK * 2);
    unsigned short* fgradb  = (unsigned short*)carve((size_t)SFK * 2);
    float*          fgrad   = (float*)         carve((size_t)SFK * 4);
    float*          scores  = (float*)         carve((size_t)SFP * 4);
    float*          tscores = (float*)         carve((size_t)SFP * 4);  // reused as tsg
    unsigned short* mappedb = (unsigned short*)carve((size_t)SFP * 2);
    float*          t1T     = (float*)         carve((size_t)SFP * 4);
    float*          ltr     = (float*)         carve((size_t)SFP * 4);  // reused as sg
    float*          u1T     = (float*)         carve((size_t)SFP * 4);  // reused as ltr_g
    unsigned short* regtpb  = (unsigned short*)carve((size_t)SFP * 2);
    float*          accs    = (float*)         carve(64 * 4);
    float* accN = accs, *accS = accs + 16, *accL = accs + 32;

    auto gs = [](int n) { return dim3((unsigned)((n + 255) / 256)); };
    const dim3 B(256);
    const dim3 g0(7, 7, 16);     // apply_filter GEMM: N=400, M=400
    const dim3 g1(36, 7, 16);    // feat_transpose GEMM: N=2304, M=400

    // one-time setup
    k_maps<<<gs(PP_), B, 0, stream>>>(wl, wm, wsp, lab, msk, sp);
    k_f32_to_bf16<<<gs(SCHW), B, 0, stream>>>(feat,  featb,  SCHW);
    k_f32_to_bf16<<<gs(SCHW), B, 0, stream>>>(tfeat, tfeatb, SCHW);
    (void)hipMemcpyAsync(filt, filt0, (size_t)SFK * 4,
                         hipMemcpyDeviceToDevice, stream);

    for (int it = 0; it < 2; ++it) {
        k_zero<<<1, 64, 0, stream>>>(accs, 48);
        k_f32_to_bf16<<<gs(SFK), B, 0, stream>>>(filt, filtb, SFK);

        // scores = apply_filter(feat, filt)
        k_gemm_bf16<0><<<g0, B, 0, stream>>>(filtb, featb, scores, nullptr, nullptr,
                                             F_, P_, K9_, 0);
        k_postscores<<<gs(SFP), B, 0, stream>>>(scores, lab, msk, sp, mappedb);
        // f_grad = feat_transpose(feat, mapped_res) + reg_w * filt
        k_gemm_bf16<1><<<g1, B, 0, stream>>>(mappedb, featb, fgrad, filt, freg,
                                             F_, K9_, P_, 1);
        // test_scores = apply_filter(test_feat, filt)
        k_gemm_bf16<0><<<g0, B, 0, stream>>>(filtb, tfeatb, tscores, nullptr, nullptr,
                                             F_, P_, K9_, 0);
        // ltr = sep4d(test_scores);  reg_tp = sep4d_t(ltr)
        k_conv_inner<<<gs(SFP), B, 0, stream>>>(tscores, t1T, nullptr, rw1, 0, 1);
        k_conv_inner<<<gs(SFP), B, 0, stream>>>(t1T, ltr, nullptr, rw2, 0, 0);
        k_conv_inner<<<gs(SFP), B, 0, stream>>>(ltr, u1T, nullptr, rw2, 1, 1);
        k_conv_inner<<<gs(SFP), B, 0, stream>>>(u1T, nullptr, regtpb, rw1, 1, 0);
        // f_grad += feat_transpose(test_feat, reg_tp)
        k_gemm_bf16<1><<<g1, B, 0, stream>>>(regtpb, tfeatb, fgrad, nullptr, nullptr,
                                             F_, K9_, P_, 2);

        k_f32_to_bf16<<<gs(SFK), B, 0, stream>>>(fgrad, fgradb, SFK);
        k_reduce_sq<<<dim3((F_ * K9_ + 255) / 256, 16), B, 0, stream>>>(
            fgrad, accN, F_ * K9_);

        // sg = spatial * mask_deriv * apply_filter(feat, f_grad); accumulate |sg|^2
        k_gemm_bf16<0><<<g0, B, 0, stream>>>(fgradb, featb, ltr, nullptr, nullptr,
                                             F_, P_, K9_, 0);
        k_sg<<<gs(SFP), B, 0, stream>>>(ltr, scores, msk, sp);
        k_reduce_sq<<<dim3((PP_ + 255) / 256, 16), B, 0, stream>>>(ltr, accS, PP_);

        // ltr_g = sep4d(apply_filter(test_feat, f_grad)); accumulate |ltr_g|^2
        k_gemm_bf16<0><<<g0, B, 0, stream>>>(fgradb, tfeatb, tscores, nullptr, nullptr,
                                             F_, P_, K9_, 0);
        k_conv_inner<<<gs(SFP), B, 0, stream>>>(tscores, t1T, nullptr, rw1, 0, 1);
        k_conv_inner<<<gs(SFP), B, 0, stream>>>(t1T, u1T, nullptr, rw2, 0, 0);
        k_reduce_sq<<<dim3((PP_ + 255) / 256, 16), B, 0, stream>>>(u1T, accL, PP_);

        // filt -= exp(lsl) * alpha[s] * f_grad
        k_update<<<gs(SFK), B, 0, stream>>>(filt, fgrad, accN, accS, accL, lsl, freg);
    }
}
